// AudioVisualContrastive_26972394619023
// MI455X (gfx1250) — matrix operand order repaired and checked by reading, not compiled, not run on hardware
//
#include <hip/hip_runtime.h>
#include <hip/hip_bf16.h>
#include <stdint.h>

typedef unsigned short u16;
typedef __attribute__((ext_vector_type(16))) __bf16 v16bf;
typedef __attribute__((ext_vector_type(8)))  float  v8f;
typedef int v4i __attribute__((vector_size(16)));   // matches builtin param type

__device__ __forceinline__ u16 f2bf(float f) {
  unsigned u = __builtin_bit_cast(unsigned, f);
  unsigned r = u + 0x7FFFu + ((u >> 16) & 1u);   // round-to-nearest-even
  return (u16)(r >> 16);
}
__device__ __forceinline__ float bf2f(u16 h) {
  unsigned u = ((unsigned)h) << 16;
  return __builtin_bit_cast(float, u);
}

union Frag { uint4 q[2]; v16bf v; };

#if __has_builtin(__builtin_amdgcn_global_load_async_to_lds_b128)
#define HAVE_ASYNC_COPY 1
#else
#define HAVE_ASYNC_COPY 0
#endif

__device__ __forceinline__ void await_async() {
#if HAVE_ASYNC_COPY
#if __has_builtin(__builtin_amdgcn_s_wait_asynccnt)
  __builtin_amdgcn_s_wait_asynccnt(0);
#else
  asm volatile("s_wait_asynccnt 0x0" ::: "memory");
#endif
#endif
}

// ---------------------------------------------------------------------------
// f32 -> bf16 elementwise convert
// ---------------------------------------------------------------------------
__global__ void k_cvt(const float* __restrict__ s, u16* __restrict__ d, size_t n) {
  size_t i = (size_t)blockIdx.x * blockDim.x + threadIdx.x;
  size_t stride = (size_t)gridDim.x * blockDim.x;
  for (; i < n; i += stride) d[i] = f2bf(s[i]);
}

// ---------------------------------------------------------------------------
// WMMA GEMM:  C[M,N] = act(A[M,K] @ W[N,K]^T + bias)
// bf16 in, f32 accumulate via v_wmma_f32_16x16x32_bf16.
// Block = 8 waves -> 128M x 64N tile. B (shared by all waves) is staged in
// LDS per 128-K chunk, double-buffered via async copies (GLOBAL_LOAD_ASYNC_
// TO_LDS_B128 / s_wait_asynccnt). A fragments ping-pong between two register
// sets (no copies, no WMMA->VALU hazard NOPs); all four B fragments are
// fetched from LDS before the WMMA burst so dscnt waits overlap compute.
// Requires M%128==0, N%64==0, K%128==0 (true at every call site here).
// ---------------------------------------------------------------------------
#define BK 128

template <int ACT, bool OUTBF>
__global__ __launch_bounds__(256) void k_gemm(const u16* __restrict__ A,
                                              const u16* __restrict__ W,
                                              const float* __restrict__ bias,
                                              void* __restrict__ Cp,
                                              int K, int ldc, int coloff) {
  constexpr int LDB = BK + 8;                 // padded LDS row stride (elems)
  __shared__ u16 Bs[2][64 * LDB];

  const int tid = threadIdx.x;
  const int lane = tid & 31;
  const int wave = tid >> 5;
  const int mb = (blockIdx.x * 8 + wave) * 16;
  const int nb = blockIdx.y * 64;
  const int lr = lane & 15, lh = lane >> 4;

  // staging: 256 threads cover 64 rows x 128 K (32 elems = 4 x 16B each)
  const int srow = tid >> 2;                  // 0..63
  const int sseg = (tid & 3) * 32;            // elem offset within chunk
  const u16* gsrc0 = W + (size_t)(nb + srow) * K + sseg;
  u16* lrow0 = &Bs[0][0] + srow * LDB + sseg;

  auto stage = [&](int buf, int kc) {
    const u16* g = gsrc0 + kc;
    u16* l = lrow0 + buf * (64 * LDB);
#pragma unroll
    for (int u = 0; u < 4; ++u) {
#if HAVE_ASYNC_COPY
      __builtin_amdgcn_global_load_async_to_lds_b128(
          (__attribute__((address_space(1))) v4i*)(g + u * 8),
          (__attribute__((address_space(3))) v4i*)(l + u * 8), 0, 0);
#else
      *(uint4*)(l + u * 8) = *(const uint4*)(g + u * 8);
#endif
    }
  };

  v8f acc[4] = {};
  // A: lane holds row (mb+lr); K-chunks [off,off+8) and [off+16,off+24), off=8*lh
  const u16* arow = A + (size_t)(mb + lr) * K + (lh << 3);

  stage(0, 0);
  Frag a[2];                                  // ping-pong A fragments
  a[0].q[0] = *(const uint4*)(arow);
  a[0].q[1] = *(const uint4*)(arow + 16);
  await_async();
  __syncthreads();

  for (int kc = 0; kc < K; kc += BK) {
    const int cur = (kc / BK) & 1;
    if (kc + BK < K) stage(cur ^ 1, kc + BK);
    const u16* bbase = &Bs[cur][0] + lr * LDB + (lh << 3);
#pragma unroll
    for (int k = 0; k < BK; k += 32) {
      const int kg = kc + k;
      const int pp = (kg >> 5) & 1;
      if (kg + 32 < K) {                      // pipeline next A into other set
        a[pp ^ 1].q[0] = *(const uint4*)(arow + kg + 32);
        a[pp ^ 1].q[1] = *(const uint4*)(arow + kg + 48);
      }
      // fetch all four B fragments before the WMMA burst
      Frag b0, b1, b2, b3;
      const u16* bp = bbase + k;
      b0.q[0] = *(const uint4*)(bp);
      b0.q[1] = *(const uint4*)(bp + 16);
      b1.q[0] = *(const uint4*)(bp + 16 * LDB);
      b1.q[1] = *(const uint4*)(bp + 16 * LDB + 16);
      b2.q[0] = *(const uint4*)(bp + 32 * LDB);
      b2.q[1] = *(const uint4*)(bp + 32 * LDB + 16);
      b3.q[0] = *(const uint4*)(bp + 48 * LDB);
      b3.q[1] = *(const uint4*)(bp + 48 * LDB + 16);
      acc[0] = __builtin_amdgcn_wmma_f32_16x16x32_bf16(
          false, a[pp].v, false, b0.v, (short)0, acc[0], false, false);
      acc[1] = __builtin_amdgcn_wmma_f32_16x16x32_bf16(
          false, a[pp].v, false, b1.v, (short)0, acc[1], false, false);
      acc[2] = __builtin_amdgcn_wmma_f32_16x16x32_bf16(
          false, a[pp].v, false, b2.v, (short)0, acc[2], false, false);
      acc[3] = __builtin_amdgcn_wmma_f32_16x16x32_bf16(
          false, a[pp].v, false, b3.v, (short)0, acc[3], false, false);
    }
    if (kc + BK < K) await_async();
    __syncthreads();
  }

  // D layout: VGPR r -> row mb + r + 8*lh, col nb + nt*16 + lr
#pragma unroll
  for (int nt = 0; nt < 4; ++nt) {
    const int col = nb + nt * 16 + lr;
    const float bv = bias[col];
#pragma unroll
    for (int r = 0; r < 8; ++r) {
      const int row = mb + r + (lh << 3);
      float v = acc[nt][r] + bv;
      if (ACT == 1) v = fmaxf(v, 0.f);
      const size_t idx = (size_t)row * ldc + coloff + col;
      if constexpr (OUTBF) ((u16*)Cp)[idx] = f2bf(v);
      else                 ((float*)Cp)[idx] = v;
    }
  }
}

// ---------------------------------------------------------------------------
// v_before[b,c] = sum_n att[b,n] * video_feat[b*36+n, c]   (b<640, c<512, n<36)
// ---------------------------------------------------------------------------
__global__ __launch_bounds__(256) void k_vbefore(const float* __restrict__ att,
                                                 const float* __restrict__ vf,
                                                 float* __restrict__ vb) {
  const int b = blockIdx.x;
  __shared__ float att_s[36];
  if (threadIdx.x < 36) att_s[threadIdx.x] = att[b * 36 + threadIdx.x];
  __syncthreads();
  for (int c = threadIdx.x; c < 512; c += 256) {
    float acc = 0.f;
    const float* base = vf + (size_t)b * 36 * 512 + c;
#pragma unroll 4
    for (int n = 0; n < 36; ++n) acc += att_s[n] * base[n * 512];
    vb[(size_t)b * 512 + c] = acc;
  }
}

// ---------------------------------------------------------------------------
// Row L2-normalize (cols=512) f32 -> bf16
// ---------------------------------------------------------------------------
__global__ __launch_bounds__(256) void k_norm(const float* __restrict__ src,
                                              u16* __restrict__ dst, int cols) {
  const int row = blockIdx.x;
  const float* s = src + (size_t)row * cols;
  float acc = 0.f;
  for (int c = threadIdx.x; c < cols; c += 256) { float v = s[c]; acc += v * v; }
  for (int o = 16; o; o >>= 1) acc += __shfl_xor(acc, o, 32);
  __shared__ float wsum[8];
  __shared__ float scale_s;
  if ((threadIdx.x & 31) == 0) wsum[threadIdx.x >> 5] = acc;
  __syncthreads();
  if (threadIdx.x == 0) {
    float t = 0.f;
    for (int i = 0; i < 8; ++i) t += wsum[i];
    scale_s = 1.f / fmaxf(sqrtf(t), 1e-12f);
  }
  __syncthreads();
  const float sc = scale_s;
  u16* d = dst + (size_t)row * cols;
  for (int c = threadIdx.x; c < cols; c += 256) d[c] = f2bf(s[c] * sc);
}

// ---------------------------------------------------------------------------
// Grounding block: one workgroup per (j,t). Stages the 36x512 vis tile in LDS
// (stride 513 -> conflict-free row access), then each wave walks i = w, w+8...
// computing scores -> softmax(36) -> grd, and writes
//   gl[(i*64+j)*10+t]   = [ tanh(v_before[j,t,:]) | tanh(grd) ]   (bf16)
//   feat[(i*64+j)*10+t] = [ audio_feat[i,t,:]     |  (gemm fills)] (bf16)
// ---------------------------------------------------------------------------
__global__ __launch_bounds__(256) void k_ground(const u16* __restrict__ vis_bf,
                                                const u16* __restrict__ aud_bf,
                                                const float* __restrict__ v_before,
                                                const float* __restrict__ audio_feat,
                                                u16* __restrict__ gl_bf,
                                                u16* __restrict__ feat_bf) {
  extern __shared__ char smem[];
  float* vtile = (float*)smem;                  // [36][513]
  float* audl  = vtile + 36 * 513;              // [8][512]
  float* parr  = audl + 8 * 512;                // [8][40]
  u16*   tb    = (u16*)(parr + 8 * 40);         // [512]

  const int jt = blockIdx.x;
  const int j = jt / 10, t = jt % 10;
  const int tid = threadIdx.x;
  const int lane = tid & 31, w = tid >> 5;
  const size_t vrow0 = (size_t)(j * 10 + t) * 36;

  for (int idx = tid; idx < 36 * 512; idx += 256) {
    int n = idx >> 9, c = idx & 511;
    vtile[n * 513 + c] = bf2f(vis_bf[(vrow0 + n) * 512 + c]);
  }
  for (int c = tid; c < 512; c += 256)
    tb[c] = f2bf(tanhf(v_before[(size_t)(j * 10 + t) * 512 + c]));
  __syncthreads();

  float* al = audl + w * 512;
  float* pw = parr + w * 40;
  const int n0 = lane, n1 = lane + 32;
  const float* v0p = vtile + n0 * 513;
  const float* v1p = vtile + (n1 < 36 ? n1 : 0) * 513;

  for (int i = w; i < 64; i += 8) {
    const size_t arow = (size_t)(i * 10 + t) * 512;
    for (int c = lane; c < 512; c += 32) al[c] = bf2f(aud_bf[arow + c]);
    // wave-private LDS region; DS ops are in-order within a wave.

    float s0 = 0.f, s1 = 0.f;
    for (int c = 0; c < 512; ++c) {
      float a = al[c];
      s0 += v0p[c] * a;
      s1 += v1p[c] * a;
    }
    if (n1 >= 36) s1 = -1e30f;
    float mx = fmaxf(s0, s1);
    for (int o = 16; o; o >>= 1) mx = fmaxf(mx, __shfl_xor(mx, o, 32));
    float e0 = __expf(s0 - mx);
    float e1 = (n1 < 36) ? __expf(s1 - mx) : 0.f;
    float sm = e0 + e1;
    for (int o = 16; o; o >>= 1) sm += __shfl_xor(sm, o, 32);
    const float inv = 1.f / sm;
    pw[n0] = e0 * inv;
    if (n1 < 36) pw[n1] = e1 * inv;

    const size_t row = (size_t)(i * 64 + j) * 10 + t;
    u16* glr = gl_bf + row * 1024;
    u16* ftr = feat_bf + row * 1024;
    const float* af = audio_feat + arow;
    for (int c = lane; c < 512; c += 32) {
      float g = 0.f;
      for (int n = 0; n < 36; ++n) g += pw[n] * vtile[n * 513 + c];
      glr[512 + c] = f2bf(tanhf(g));
      glr[c] = tb[c];
      ftr[c] = f2bf(af[c]);
    }
  }
}

// ---------------------------------------------------------------------------
// Final layer: out[r] = sum_c feat3[r,c]*W4[c] + b4   (K=128, N=1)
// ---------------------------------------------------------------------------
__global__ __launch_bounds__(256) void k_last(const u16* __restrict__ f3,
                                              const float* __restrict__ W4,
                                              const float* __restrict__ b4,
                                              float* __restrict__ out, int rows) {
  const int lane = threadIdx.x & 31;
  const int row = (blockIdx.x * 256 + threadIdx.x) >> 5;
  if (row >= rows) return;
  float acc = 0.f;
  for (int c = lane; c < 128; c += 32) acc += bf2f(f3[(size_t)row * 128 + c]) * W4[c];
  for (int o = 16; o; o >>= 1) acc += __shfl_xor(acc, o, 32);
  if (lane == 0) out[row] = acc + b4[0];
}

// ---------------------------------------------------------------------------
extern "C" void kernel_launch(void* const* d_in, const int* in_sizes, int n_in,
                              void* d_out, int out_size, void* d_ws, size_t ws_size,
                              hipStream_t stream) {
  (void)in_sizes; (void)n_in; (void)out_size; (void)ws_size;
  const float* video = (const float*)d_in[0];   // (640,36,1536)
  const float* audio = (const float*)d_in[1];   // (640,1,768)
  const float* att   = (const float*)d_in[2];   // (640,1,36)
  const float* Wv1   = (const float*)d_in[3];
  const float* bv1   = (const float*)d_in[4];
  const float* Wa1   = (const float*)d_in[5];
  const float* ba1   = (const float*)d_in[6];
  const float* Wgl   = (const float*)d_in[7];
  const float* bgl   = (const float*)d_in[8];
  const float* W1    = (const float*)d_in[9];
  const float* b1    = (const float*)d_in[10];
  const float* W2    = (const float*)d_in[11];
  const float* b2    = (const float*)d_in[12];
  const float* W3    = (const float*)d_in[13];
  const float* b3    = (const float*)d_in[14];
  const float* W4    = (const float*)d_in[15];
  const float* b4    = (const float*)d_in[16];
  float* out = (float*)d_out;

  char* ws = (char*)d_ws;
  size_t off = 0;
  auto take = [&](size_t bytes) -> char* {
    char* p = ws + off;
    off += (bytes + 255) & ~(size_t)255;
    return p;
  };

  u16*   wv1b   = (u16*)  take((size_t)512 * 1536 * 2);
  u16*   wa1b   = (u16*)  take((size_t)512 * 768 * 2);
  u16*   wglb   = (u16*)  take((size_t)512 * 1024 * 2);
  u16*   w1b    = (u16*)  take((size_t)512 * 1024 * 2);
  u16*   w2b    = (u16*)  take((size_t)256 * 512 * 2);
  u16*   w3b    = (u16*)  take((size_t)128 * 256 * 2);
  u16*   vidb   = (u16*)  take((size_t)23040 * 1536 * 2);
  u16*   audb   = (u16*)  take((size_t)640 * 768 * 2);
  float* vfeat  = (float*)take((size_t)23040 * 512 * 4);
  float* afeat  = (float*)take((size_t)640 * 512 * 4);
  float* vbefore= (float*)take((size_t)640 * 512 * 4);
  u16*   visb   = (u16*)  take((size_t)23040 * 512 * 2);
  u16*   audnb  = (u16*)  take((size_t)640 * 512 * 2);
  u16*   glb    = (u16*)  take((size_t)40960 * 1024 * 2);
  u16*   featb  = (u16*)  take((size_t)40960 * 1024 * 2);
  u16*   f1b    = (u16*)  take((size_t)40960 * 512 * 2);
  u16*   f2b    = (u16*)  take((size_t)40960 * 256 * 2);
  u16*   f3b    = (u16*)  take((size_t)40960 * 128 * 2);

  auto cvt = [&](const float* s, u16* d, size_t n) {
    int blocks = (int)((n + 1023) / 1024);
    if (blocks < 1) blocks = 1;
    k_cvt<<<blocks, 256, 0, stream>>>(s, d, n);
  };
  cvt(Wv1, wv1b, (size_t)512 * 1536);
  cvt(Wa1, wa1b, (size_t)512 * 768);
  cvt(Wgl, wglb, (size_t)512 * 1024);
  cvt(W1,  w1b,  (size_t)512 * 1024);
  cvt(W2,  w2b,  (size_t)256 * 512);
  cvt(W3,  w3b,  (size_t)128 * 256);
  cvt(video, vidb, (size_t)23040 * 1536);
  cvt(audio, audb, (size_t)640 * 768);

  // video_feat = video @ Wv1^T + bv1  (f32 out, needed for norms & v_before)
  k_gemm<0, false><<<dim3(23040 / 128, 512 / 64), 256, 0, stream>>>(
      vidb, wv1b, bv1, vfeat, 1536, 512, 0);
  // audio_feat = audio @ Wa1^T + ba1
  k_gemm<0, false><<<dim3(640 / 128, 512 / 64), 256, 0, stream>>>(
      audb, wa1b, ba1, afeat, 768, 512, 0);

  k_vbefore<<<640, 256, 0, stream>>>(att, vfeat, vbefore);
  k_norm<<<23040, 256, 0, stream>>>(vfeat, visb, 512);
  k_norm<<<640, 256, 0, stream>>>(afeat, audnb, 512);

  const int GSMEM = (36 * 513 + 8 * 512 + 8 * 40) * 4 + 512 * 2;  // ~92.6 KB
  k_ground<<<640, 256, GSMEM, stream>>>(visb, audnb, vbefore, afeat, glb, featb);

  // visual_feat_grd = tanh(gl) @ Wgl^T + bgl  -> feat[:, 512:1024] (bf16)
  k_gemm<0, true><<<dim3(40960 / 128, 512 / 64), 256, 0, stream>>>(
      glb, wglb, bgl, featb, 1024, 1024, 512);
  // feat1 = relu(feat @ W1^T + b1)
  k_gemm<1, true><<<dim3(40960 / 128, 512 / 64), 256, 0, stream>>>(
      featb, w1b, b1, f1b, 1024, 512, 0);
  // feat2 = relu(feat1 @ W2^T + b2)
  k_gemm<1, true><<<dim3(40960 / 128, 256 / 64), 256, 0, stream>>>(
      f1b, w2b, b2, f2b, 512, 256, 0);
  // feat3 = relu(feat2 @ W3^T + b3)
  k_gemm<1, true><<<dim3(40960 / 128, 128 / 64), 256, 0, stream>>>(
      f2b, w3b, b3, f3b, 256, 128, 0);
  // out = feat3 @ W4^T + b4
  k_last<<<40960 / 8, 256, 0, stream>>>(f3b, W4, b4, out, 40960);
}
